// BSNNet_515396076248
// MI455X (gfx1250) — matrix operand order repaired and checked by reading, not compiled, not run on hardware
//
#include <hip/hip_runtime.h>
#include <hip/hip_bf16.h>

typedef _Float16 v16h __attribute__((ext_vector_type(16)));
typedef _Float16 v8h  __attribute__((ext_vector_type(8)));
typedef float    v8f  __attribute__((ext_vector_type(8)));

#define BN   64
#define TT   100
#define FD   400
#define CHID 512
#define C3   1003
#define KP   500

#define TPAD 128   // padded time rows; row 0 <=> t = -1
#define CIN1 416   // FEAT_DIM padded to multiple of 32
#define MT   7     // time tiles of 16 (112 >= 100)

// ---------------- WMMA fragment loaders (contiguous k order: lane half h holds k 16h..16h+15) -----

static __device__ __forceinline__ v16h load_a_frag(const _Float16* rowptr, int k0, int half) {
  // A 16x32 f16: lo-half lanes: K k0+0..15 ; hi-half: K k0+16..31
  v8h lo = *reinterpret_cast<const v8h*>(rowptr + k0 + 16 * half);
  v8h hi = *reinterpret_cast<const v8h*>(rowptr + k0 + 16 * half + 8);
  v16h a;
#pragma unroll
  for (int i = 0; i < 8; ++i) { a[i] = lo[i]; a[8 + i] = hi[i]; }
  return a;
}

static __device__ __forceinline__ v16h load_b_frag(const _Float16* p) {
  // packed tile: lane's 16 K-halves are contiguous (32 bytes)
  v8h lo = *reinterpret_cast<const v8h*>(p);
  v8h hi = *reinterpret_cast<const v8h*>(p + 8);
  v16h b;
#pragma unroll
  for (int i = 0; i < 8; ++i) { b[i] = lo[i]; b[8 + i] = hi[i]; }
  return b;
}

// ---------------- prep kernels --------------------------------------------

__global__ void prep_x_kernel(const float* __restrict__ x, _Float16* __restrict__ xTm) {
  int tid = blockIdx.x * 256 + threadIdx.x;
  const int total = BN * TPAD * CIN1;
  if (tid >= total) return;
  int c  = tid % CIN1;
  int tp = (tid / CIN1) % TPAD;
  int b  = tid / (CIN1 * TPAD);
  int t  = tp - 1;
  float v = 0.f;
  if (t >= 0 && t < TT && c < FD) v = x[((size_t)b * TT + t) * FD + c];
  xTm[tid] = (_Float16)v;
}

__global__ void prep_w1_kernel(const float* __restrict__ w, _Float16* __restrict__ wp) {
  int tid = blockIdx.x * 256 + threadIdx.x;
  const int total = 3 * CHID * CIN1;
  if (tid >= total) return;
  int ci = tid % CIN1;
  int o  = (tid / CIN1) % CHID;
  int kk = tid / (CIN1 * CHID);
  float v = (ci < FD) ? w[(size_t)o * FD * 3 + (size_t)ci * 3 + kk] : 0.f;
  int kt = ci >> 4, kr = ci & 15, ntl = o >> 4, n = o & 15;
  wp[(((size_t)kk * 26 + kt) * 32 + ntl) * 256 + n * 16 + kr] = (_Float16)v;
}

__global__ void prep_w2_kernel(const float* __restrict__ w, _Float16* __restrict__ wp) {
  int tid = blockIdx.x * 256 + threadIdx.x;
  const int total = 3 * CHID * CHID;
  if (tid >= total) return;
  int ci = tid % CHID;
  int o  = (tid / CHID) % CHID;
  int kk = tid / (CHID * CHID);
  float v = w[(size_t)o * CHID * 3 + (size_t)ci * 3 + kk];
  int kt = ci >> 4, kr = ci & 15, ntl = o >> 4, n = o & 15;
  wp[(((size_t)kk * 32 + kt) * 32 + ntl) * 256 + n * 16 + kr] = (_Float16)v;
}

__global__ void prep_w3_kernel(const float* __restrict__ w, _Float16* __restrict__ wp) {
  int tid = blockIdx.x * 256 + threadIdx.x;
  const int total = 32 * 63 * 256;
  if (tid >= total) return;
  int within = tid % 256;
  int n = within >> 4, kr = within & 15;
  int ntl = (tid / 256) % 63;
  int kt  = tid / (256 * 63);
  int o  = ntl * 16 + n;
  int ci = kt * 16 + kr;
  float v = (o < C3) ? w[(size_t)o * CHID + ci] : 0.f;
  wp[tid] = (_Float16)v;
}

__global__ void prep_f1_kernel(const float* __restrict__ w, _Float16* __restrict__ wp) {
  int tid = blockIdx.x * 256 + threadIdx.x;
  const int total = 2 * 32 * 256;
  if (tid >= total) return;
  int within = tid % 256;
  int n = within >> 4, kr = within & 15;
  int ntl = (tid / 256) % 32;
  int kt  = tid / (256 * 32);
  int o = ntl * 16 + n;
  int k = kt * 16 + kr;
  wp[tid] = (_Float16)w[(size_t)o * 32 + k];
}

// ---------------- WMMA conv: LDS-staged A, multi-accumulator over N -------
// One block per (b, time-tile). 18 padded activation rows staged in LDS and
// re-used by all N tiles; each wave holds NACC accumulators (n-tiles
// wid, wid+8, ...), so each LDS A fragment feeds NACC WMMAs.
template <int TAPS, int NACC>
__global__ void conv_wmma_kernel(const _Float16* __restrict__ in,
                                 const _Float16* __restrict__ wp,
                                 const float* __restrict__ bias,
                                 _Float16* __restrict__ outH,
                                 float* __restrict__ outF,
                                 int ldA, int KT, int NT, int mode, int Cout)
{
  __shared__ _Float16 smem[18 * CHID];  // max 18 rows x 512 halves = 18 KB
  const int b  = blockIdx.x / MT;
  const int mt = blockIdx.x % MT;
  const int t0 = mt * 16;
  const int lane = threadIdx.x & 31;
  const int wid  = threadIdx.x >> 5;
  const int row = lane & 15, half = lane >> 4;

  // stage padded rows t0 .. t0+17 into LDS (16-byte units, coalesced)
  const _Float16* inB = in + (size_t)b * TPAD * ldA;
  const int units = 18 * (ldA >> 3);
  for (int u = threadIdx.x; u < units; u += 256) {
    const int r  = u / (ldA >> 3);
    const int cc = u - r * (ldA >> 3);
    *reinterpret_cast<v8h*>(&smem[r * ldA + cc * 8]) =
        *reinterpret_cast<const v8h*>(inB + (size_t)(t0 + r) * ldA + cc * 8);
  }
  __syncthreads();

  v8f acc[NACC];
#pragma unroll
  for (int i = 0; i < NACC; ++i) acc[i] = (v8f){0.f, 0.f, 0.f, 0.f, 0.f, 0.f, 0.f, 0.f};

  const int kc = KT >> 1;
  for (int kk = 0; kk < TAPS; ++kk) {
    const int lr = (TAPS == 3 ? kk : 1) + row;  // local LDS row of this lane's M row
    const _Float16* arow = smem + (size_t)lr * ldA;
    for (int c2 = 0; c2 < kc; ++c2) {
      const int k0 = c2 * 32;
      v16h a = load_a_frag(arow, k0, half);  // ds_load_b128 x2
      const size_t kbase = (size_t)(kk * KT + c2 * 2 + half) * NT;
#pragma unroll
      for (int i = 0; i < NACC; ++i) {
        const int nt = wid + 8 * i;          // wave-uniform
        if (nt < NT) {
          const _Float16* bp = wp + (kbase + nt) * 256 + row * 16;
          if (c2 + 1 < kc) __builtin_prefetch(bp + 2 * (size_t)NT * 256, 0, 1);
          v16h bf = load_b_frag(bp);
          acc[i] = __builtin_amdgcn_wmma_f32_16x16x32_f16(false, a, false, bf, (short)0,
                                                          acc[i], false, false);
        }
      }
    }
  }

#pragma unroll
  for (int i = 0; i < NACC; ++i) {
    const int nt = wid + 8 * i;
    if (nt >= NT) continue;
    const int o = nt * 16 + row;
    if (mode == 0) {  // relu -> f16 activation store (time-major, padded)
      const float bo = bias[o];
#pragma unroll
      for (int r = 0; r < 8; ++r) {
        const int t = t0 + half * 8 + r;
        if (t < TT) {
          float v = acc[i][r] + bo;
          v = v > 0.f ? v : 0.f;
          outH[(size_t)b * TPAD * Cout + (size_t)(t + 1) * Cout + o] = (_Float16)v;
        }
      }
    } else {  // sigmoid(0.01*(conv+b)) -> tem (channel-major f32)
      if (o < Cout) {
        const float bo = bias[o];
#pragma unroll
        for (int r = 0; r < 8; ++r) {
          const int t = t0 + half * 8 + r;
          if (t < TT) {
            const float z = 0.01f * (acc[i][r] + bo);
            outF[(size_t)b * Cout * TT + (size_t)o * TT + t] = 1.f / (1.f + __expf(-z));
          }
        }
      }
    }
  }
}

// ---------------- peaks + exact top-k by rank counting --------------------

static __device__ __forceinline__ unsigned int fkey(float f) {
  unsigned int u = __float_as_uint(f);
  return (u & 0x80000000u) ? ~u : (u | 0x80000000u);  // monotone order-preserving
}

__global__ void peaks_topk_kernel(const float* __restrict__ tem,
                                  int* __restrict__ pe, int* __restrict__ ps,
                                  float* __restrict__ okf,
                                  float* __restrict__ props, float* __restrict__ okout)
{
  __shared__ float ss[TT], ee[TT];
  __shared__ int   sm[TT], em[TT];
  __shared__ unsigned int keys[TT * TT];
  const int b = blockIdx.x, tid = threadIdx.x;
  const float* tb = tem + (size_t)b * (C3 * TT);
  if (tid < TT) { ss[tid] = tb[TT + tid]; ee[tid] = tb[2 * TT + tid]; }
  __syncthreads();
  if (tid < TT) {
    bool si = (tid == 0 || tid == TT - 1)
                  ? true
                  : (ss[tid] > ss[tid - 1] && ss[tid] > ss[tid + 1] && ss[tid] > 0.5f);
    bool ei = (tid == 0 || tid == TT - 1)
                  ? true
                  : (ee[tid] > ee[tid - 1] && ee[tid] > ee[tid + 1] && ee[tid] > 0.5f);
    sm[tid] = si; em[tid] = ei;
  }
  __syncthreads();
  for (int i = tid; i < TT * TT; i += blockDim.x) {
    int eidx = i / TT, sidx = i % TT;
    bool valid = em[eidx] && sm[sidx] && (sidx < eidx);
    float v = valid ? ee[eidx] * ss[sidx] : -1.0f;
    keys[i] = fkey(v);
  }
  __syncthreads();
  for (int i = tid; i < TT * TT; i += blockDim.x) {
    const unsigned int ki = keys[i];
    int rank = 0;
    for (int j = 0; j < TT * TT; ++j) {
      const unsigned int kj = keys[j];
      rank += (kj > ki) || (kj == ki && j < i);
    }
    if (rank < KP) {
      const int bp = b * KP + rank;
      const int eidx = i / TT, sidx = i % TT;
      pe[bp] = eidx; ps[bp] = sidx;
      const float okv = (ki >= 0x80000000u) ? 1.f : 0.f;  // value >= 0
      okf[bp] = okv;
      props[(size_t)bp * 2 + 0] = (float)eidx;
      props[(size_t)bp * 2 + 1] = (float)sidx;
      okout[bp] = okv;
    }
  }
}

// ---------------- logits via per-(b,c) prefix sums ------------------------

__global__ void logits_kernel(const float* __restrict__ tem,
                              const int* __restrict__ pe, const int* __restrict__ ps,
                              const float* __restrict__ okf, float* __restrict__ lg)
{
  const int b = blockIdx.x;
  const int c = blockIdx.y * 256 + threadIdx.x;
  if (c >= 1000) return;
  const float* row = tem + (size_t)b * (C3 * TT) + (size_t)(3 + c) * TT;
  float cs[TT + 1];
  cs[0] = 0.f;
  for (int t = 0; t < TT; ++t) cs[t + 1] = cs[t] + row[t];
  for (int p = 0; p < KP; ++p) {
    const int bp = b * KP + p;
    const int e = pe[bp], s = ps[bp];
    const float ok = okf[bp];
    int d = e - s; if (d < 1) d = 1;
    lg[(size_t)bp * 1000 + c] = ok * (cs[e] - cs[s]) / (float)d;
  }
}

// ---------------- proposal interpolation features -------------------------

static __device__ __forceinline__ float interp_apad(const float* a0, float pos) {
  float p = pos * (170.0f / 169.0f) - 0.5f;
  float x0f = floorf(p);
  int x0 = (int)x0f;
  float w = p - x0f;
  float v0 = 0.f, v1 = 0.f;
  if (x0 >= 0 && x0 < 170) { int j = x0 - 35; if (j >= 0 && j < TT) v0 = a0[j]; }
  int x1 = x0 + 1;
  if (x1 >= 0 && x1 < 170) { int j = x1 - 35; if (j >= 0 && j < TT) v1 = a0[j]; }
  return v0 * (1.f - w) + v1 * w;
}

__global__ void feat_kernel(const float* __restrict__ tem,
                            const int* __restrict__ pe, const int* __restrict__ ps,
                            const float* __restrict__ okf, _Float16* __restrict__ featp)
{
  __shared__ float a0[TT];
  const int b = blockIdx.x;
  const int p = blockIdx.y * 256 + threadIdx.x;
  if (threadIdx.x < TT) a0[threadIdx.x] = tem[(size_t)b * (C3 * TT) + threadIdx.x];
  __syncthreads();
  if (p >= KP) return;
  const int bp = b * KP + p;
  const int e = pe[bp], s = ps[bp];
  const float ok = okf[bp];
  const float xmin = (float)e, xmax = (float)s;
  const float xlen = xmax - xmin;
  const float plen_b = 2.0f * 0.2f * xlen / 7.0f;
  const float plen_a = xlen / 15.0f;
  const float base_s = xmin - 0.2f * xlen - plen_b * 0.5f;
  const float base_e = xmax - 0.2f * xlen - plen_b * 0.5f;
  const float base_a = xmin - plen_a * 0.5f;
  const float step_b = plen_b / 3.0f;
  const float step_a = plen_a / 3.0f;
  float feat[32];
  for (int g = 0; g < 16; ++g) {
    float sum = 0.f;
    for (int j = 0; j < 4; ++j) sum += interp_apad(a0, base_a + step_a * (float)(3 * g + j));
    feat[g] = sum * 0.25f;
  }
  for (int g = 0; g < 8; ++g) {
    float sum = 0.f;
    for (int j = 0; j < 4; ++j) sum += interp_apad(a0, base_s + step_b * (float)(3 * g + j));
    feat[16 + g] = sum * 0.25f;
  }
  for (int g = 0; g < 8; ++g) {
    float sum = 0.f;
    for (int j = 0; j < 4; ++j) sum += interp_apad(a0, base_e + step_b * (float)(3 * g + j));
    feat[24 + g] = sum * 0.25f;
  }
  for (int f = 0; f < 32; ++f) featp[(size_t)bp * 32 + f] = (_Float16)(feat[f] * ok);
}

// ---------------- FC (feat @ f1w.T) via WMMA, fused pem -------------------

__global__ void fc_pem_kernel(const _Float16* __restrict__ featp,
                              const _Float16* __restrict__ f1p,
                              const float* __restrict__ f1b,
                              const float* __restrict__ f2w,
                              const float* __restrict__ f2b,
                              const float* __restrict__ okf,
                              float* __restrict__ pem)
{
  const int lane = threadIdx.x & 31;
  const int wid  = threadIdx.x >> 5;
  const int gwid = blockIdx.x * (blockDim.x >> 5) + wid;
  if (gwid >= 2000) return;           // 32000 proposals / 16
  const int m0 = gwid * 16;
  const int row = lane & 15, half = lane >> 4;
  const _Float16* arow = featp + (size_t)(m0 + row) * 32;
  v16h a = load_a_frag(arow, 0, half);  // K = 32 exactly
  float part[8] = {0.f, 0.f, 0.f, 0.f, 0.f, 0.f, 0.f, 0.f};
  for (int nt = 0; nt < 32; ++nt) {
    const _Float16* bp = f1p + ((size_t)(half * 32 + nt)) * 256 + row * 16;
    v16h bf = load_b_frag(bp);
    v8f acc = {0.f, 0.f, 0.f, 0.f, 0.f, 0.f, 0.f, 0.f};
    acc = __builtin_amdgcn_wmma_f32_16x16x32_f16(false, a, false, bf, (short)0, acc,
                                                 false, false);
    const int o = nt * 16 + row;
    const float b1 = f1b[o];
    const float w2 = f2w[o];
#pragma unroll
    for (int r = 0; r < 8; ++r) {
      float h = 0.1f * (acc[r] + b1);
      h = h > 0.f ? h : 0.f;
      part[r] += h * w2;
    }
  }
  // reduce across the 16 lanes of each half (xor masks < 16 stay within half)
#pragma unroll
  for (int off = 1; off < 16; off <<= 1) {
#pragma unroll
    for (int r = 0; r < 8; ++r) part[r] += __shfl_xor(part[r], off, 32);
  }
  if (row == 0) {
#pragma unroll
    for (int r = 0; r < 8; ++r) {
      const int p = m0 + half * 8 + r;
      const float z = 0.1f * (part[r] + f2b[0]);
      pem[p] = (1.f / (1.f + __expf(-z))) * okf[p];
    }
  }
}

// ---------------- host side -----------------------------------------------

extern "C" void kernel_launch(void* const* d_in, const int* in_sizes, int n_in,
                              void* d_out, int out_size, void* d_ws, size_t ws_size,
                              hipStream_t stream)
{
  const float* x   = (const float*)d_in[0];
  const float* c1w = (const float*)d_in[1];
  const float* c1b = (const float*)d_in[2];
  const float* c2w = (const float*)d_in[3];
  const float* c2b = (const float*)d_in[4];
  const float* c3w = (const float*)d_in[5];
  const float* c3b = (const float*)d_in[6];
  const float* f1w = (const float*)d_in[7];
  const float* f1b = (const float*)d_in[8];
  const float* f2w = (const float*)d_in[9];
  const float* f2b = (const float*)d_in[10];

  char* ws = (char*)d_ws;
  size_t off = 0;
  _Float16* xTm   = (_Float16*)(ws + off); off += (size_t)BN * TPAD * CIN1 * 2;
  _Float16* h1    = (_Float16*)(ws + off); off += (size_t)BN * TPAD * CHID * 2;
  _Float16* h2    = (_Float16*)(ws + off); off += (size_t)BN * TPAD * CHID * 2;
  _Float16* w1p   = (_Float16*)(ws + off); off += (size_t)3 * 26 * 32 * 256 * 2;
  _Float16* w2p   = (_Float16*)(ws + off); off += (size_t)3 * 32 * 32 * 256 * 2;
  _Float16* w3p   = (_Float16*)(ws + off); off += (size_t)32 * 63 * 256 * 2;
  _Float16* f1p   = (_Float16*)(ws + off); off += (size_t)2 * 32 * 256 * 2;
  _Float16* featp = (_Float16*)(ws + off); off += (size_t)BN * KP * 32 * 2;
  int*      pe    = (int*)(ws + off);      off += (size_t)BN * KP * 4;
  int*      ps    = (int*)(ws + off);      off += (size_t)BN * KP * 4;
  float*    okf   = (float*)(ws + off);    off += (size_t)BN * KP * 4;

  float* out    = (float*)d_out;
  float* tem    = out;                          // 64*1003*100
  float* pem    = tem + (size_t)BN * C3 * TT;   // 64*500
  float* logits = pem + (size_t)BN * KP;        // 64*500*1000
  float* props  = logits + (size_t)BN * KP * 1000;  // 64*500*2
  float* okout  = props + (size_t)BN * KP * 2;  // 64*500

  // zero activation buffers (halo rows must be zero)
  hipMemsetAsync(h1, 0, (size_t)BN * TPAD * CHID * 2, stream);
  hipMemsetAsync(h2, 0, (size_t)BN * TPAD * CHID * 2, stream);

  {
    int n = BN * TPAD * CIN1;
    prep_x_kernel<<<(n + 255) / 256, 256, 0, stream>>>(x, xTm);
  }
  { int n = 3 * CHID * CIN1;  prep_w1_kernel<<<(n + 255) / 256, 256, 0, stream>>>(c1w, w1p); }
  { int n = 3 * CHID * CHID;  prep_w2_kernel<<<(n + 255) / 256, 256, 0, stream>>>(c2w, w2p); }
  { int n = 32 * 63 * 256;    prep_w3_kernel<<<(n + 255) / 256, 256, 0, stream>>>(c3w, w3p); }
  { int n = 2 * 32 * 256;     prep_f1_kernel<<<(n + 255) / 256, 256, 0, stream>>>(f1w, f1p); }

  // one block per (b, time-tile); 8 waves; NACC accumulators per wave
  // conv1: K=416 (26 ktiles), N=512 (32 ntiles = 8 waves x 4), 3 taps
  conv_wmma_kernel<3, 4><<<BN * MT, 256, 0, stream>>>(
      xTm, w1p, c1b, h1, nullptr, CIN1, 26, 32, 0, CHID);
  // conv2: K=512, N=512, 3 taps
  conv_wmma_kernel<3, 4><<<BN * MT, 256, 0, stream>>>(
      h1, w2p, c2b, h2, nullptr, CHID, 32, 32, 0, CHID);
  // conv3: K=512, N=1008 (63 ntiles = 8 waves x 8, last guarded), 1 tap
  conv_wmma_kernel<1, 8><<<BN * MT, 256, 0, stream>>>(
      h2, w3p, c3b, nullptr, tem, CHID, 32, 63, 1, C3);

  peaks_topk_kernel<<<BN, 512, 0, stream>>>(tem, pe, ps, okf, props, okout);
  logits_kernel<<<dim3(BN, 4), 256, 0, stream>>>(tem, pe, ps, okf, logits);
  feat_kernel<<<dim3(BN, 2), 256, 0, stream>>>(tem, pe, ps, okf, featp);
  fc_pem_kernel<<<(2000) / 8, 256, 0, stream>>>(featp, f1p, f1b, f2w, f2b, okf, pem);

  (void)in_sizes; (void)n_in; (void)out_size; (void)ws_size;
}
